// BasicTransformerBlock_12738873000028
// MI455X (gfx1250) — matrix operand order repaired and checked by reading, not compiled, not run on hardware
//
#include <hip/hip_runtime.h>
#include <hip/hip_bf16.h>
#include <cmath>

// ---------------------------------------------------------------------------
// BasicTransformerBlock for MI455X (gfx1250), bf16 WMMA pipeline, v5:
//  - double-buffered LDS, software-pipelined k-loop, running-pointer staging
//  - sched_barrier keeps next-tile global loads issued ahead of the WMMAs
//  - float4 global staging; packed v_cvt_pk_bf16_f32 everywhere
//  - non-trans B staged as 2x4 micro-tiles -> full cvt_pk + ds_store_b32
//  - branch-free templated epilogues (bias / residual fusion)
//   B=2, H=W=64 -> N=4096, C=320.  All GEMMs via v_wmma_f32_16x16x32_bf16.
// ---------------------------------------------------------------------------

typedef __attribute__((ext_vector_type(16))) __bf16 v16bf;
typedef __attribute__((ext_vector_type(2)))  __bf16 v2bf;
typedef __attribute__((ext_vector_type(2)))  float  v2f;
typedef __attribute__((ext_vector_type(8)))  float  v8f;

#define BDIM 256
#define BM   128
#define BN   64
#define BK   32
#define AREG ((BM / 16) * 512)   // ushorts per A buffer (8 KB)
#define BREG ((BN / 16) * 512)   // ushorts per B buffer (4 KB)

// fp32 -> bf16 (RNE) through the hardware convert path (v_cvt_pk_bf16_f32).
__device__ __forceinline__ unsigned cvt2bf(float a, float b) {
    v2f t; t.x = a; t.y = b;
    v2bf r = __builtin_convertvector(t, v2bf);
    return __builtin_bit_cast(unsigned, r);
}

// ---------------------------------------------------------------------------
// Tiled GEMM: C[M,N] = A[M,K] @ B[K,N] (+bias +res1*s1 +res2), fp32 in/out,
// bf16 WMMA compute.  LDS tiles are stored pre-swizzled in CDNA5 fragment
// order (lanes per ISA 7.12.2; k = 16*half + e) so each lane reads its v16bf fragment as 32 contiguous
// bytes (2 x ds_load_b128).
//   A-frag (16x32, MxK): lane = m + 16*half ; elem e -> K = e + 16*half
//   B-frag (32x16, KxN): lane = n + 16*half ; elem e -> K = e + 16*half
// RES: 0 = none, 1 = +res1*res1_scale, 2 = +res1*res1_scale +res2
// ---------------------------------------------------------------------------
template <bool TRANSB, bool HAS_BIAS, int RES>
__global__ __launch_bounds__(BDIM)
void gemm_bf16_wmma(const float* __restrict__ A, const float* __restrict__ Bsrc,
                    const float* __restrict__ bias,
                    const float* __restrict__ res1, float res1_scale,
                    const float* __restrict__ res2,
                    float* __restrict__ Cout,
                    int K, int lda, int ldb, int ldc,
                    long long strideA, long long strideB, long long strideC)
{
    __shared__ __align__(16) unsigned short ldsA[2][AREG];
    __shared__ __align__(16) unsigned short ldsB[2][BREG];

    const int tid  = threadIdx.x;
    const int lane = tid & 31;
    const int wave = tid >> 5;
    const int wm   = wave >> 1;   // 0..3 : 32-row strip
    const int wn   = wave & 1;    // 0..1 : 32-col strip
    const int half = lane >> 4;
    const int l16  = lane & 15;

    const int n0 = blockIdx.x * BN;
    const int m0 = blockIdx.y * BM;
    const float* __restrict__ Ab = A + (long long)blockIdx.z * strideA;
    const float* __restrict__ Bb = Bsrc + (long long)blockIdx.z * strideB;
    float* __restrict__ Cb = Cout + (long long)blockIdx.z * strideC;

    const v8f vzero = {0.f, 0.f, 0.f, 0.f, 0.f, 0.f, 0.f, 0.f};
    v8f acc[2][2];
    acc[0][0] = vzero; acc[0][1] = vzero; acc[1][0] = vzero; acc[1][1] = vzero;

    float4 pfA[4];   // 128x32 A tile: 1024 float4 quads, 4 per thread
    float4 pfB[2];   // 32x64  B tile: 2 float4 per thread (quad or 2x4 tile)

    // ---- running global pointers (advance by one k-step per stage) --------
    const float* arun[4];
#pragma unroll
    for (int i = 0; i < 4; ++i) {
        int qi = tid + BDIM * i;
        int m  = qi >> 3;
        int kq = (qi & 7) * 4;
        arun[i] = &Ab[(long long)(m0 + m) * lda + kq];
    }
    const float* brun[2];
    if (TRANSB) {              // Bsrc is [N,K] row-major: quads along K
#pragma unroll
        for (int i = 0; i < 2; ++i) {
            int qi = tid + BDIM * i;
            int n  = qi >> 3;
            int kq = (qi & 7) * 4;
            brun[i] = &Bb[(long long)(n0 + n) * ldb + kq];
        }
    } else {                   // Bsrc is [K,N]: 2(k) x 4(n) micro-tile/thread
        int kp = tid >> 4;            // 0..15 -> k rows 2kp, 2kp+1
        int nq = (tid & 15) * 4;
        brun[0] = &Bb[(long long)(2 * kp) * ldb + (n0 + nq)];
        brun[1] = brun[0] + ldb;
    }
    const long long bstep = TRANSB ? (long long)BK : (long long)BK * ldb;

    auto loadA = [&]() {
#pragma unroll
        for (int i = 0; i < 4; ++i) { pfA[i] = *(const float4*)arun[i]; arun[i] += BK; }
    };
    auto loadB = [&]() {
        pfB[0] = *(const float4*)brun[0];
        pfB[1] = *(const float4*)brun[1];
        brun[0] += bstep; brun[1] += bstep;
    };
    // ---- convert + store prefetch registers into LDS fragment image -------
    auto storeA = [&](unsigned short* bufA) {
#pragma unroll
        for (int i = 0; i < 4; ++i) {
            int qi = tid + BDIM * i;
            int m  = qi >> 3;
            int kq = (qi & 7) * 4;
            int t = m >> 4, mm = m & 15;
            int h  = kq >> 4;
            int eb = kq & 15;
            uint2 p;
            p.x = cvt2bf(pfA[i].x, pfA[i].y);
            p.y = cvt2bf(pfA[i].z, pfA[i].w);
            *(uint2*)&bufA[t * 512 + (mm + 16 * h) * 16 + eb] = p;
        }
    };
    auto storeB = [&](unsigned short* bufB) {
        if (TRANSB) {
#pragma unroll
            for (int i = 0; i < 2; ++i) {
                int qi = tid + BDIM * i;
                int n  = qi >> 3;
                int kq = (qi & 7) * 4;
                int u = n >> 4, nn = n & 15;
                int h = kq >> 4, eb = kq & 15;
                uint2 p;
                p.x = cvt2bf(pfB[i].x, pfB[i].y);
                p.y = cvt2bf(pfB[i].z, pfB[i].w);
                *(uint2*)&bufB[u * 512 + (nn + 16 * h) * 16 + eb] = p;
            }
        } else {
            // vertical pack: (k,n) and (k+1,n) are adjacent e-slots
            int kp = tid >> 4;
            int nq = (tid & 15) * 4;
            int k0 = 2 * kp;
            int h = k0 >> 4, e0 = k0 & 15;
            int u = nq >> 4, nn = nq & 15;
            unsigned short* base = &bufB[u * 512 + (nn + 16 * h) * 16 + e0];
            *(unsigned*)(base + 0)  = cvt2bf(pfB[0].x, pfB[1].x);
            *(unsigned*)(base + 16) = cvt2bf(pfB[0].y, pfB[1].y);
            *(unsigned*)(base + 32) = cvt2bf(pfB[0].z, pfB[1].z);
            *(unsigned*)(base + 48) = cvt2bf(pfB[0].w, pfB[1].w);
        }
    };
    auto wmma_step = [&](const unsigned short* bufA, const unsigned short* bufB) {
        v16bf a0 = *(const v16bf*)&bufA[(2 * wm + 0) * 512 + lane * 16];
        v16bf a1 = *(const v16bf*)&bufA[(2 * wm + 1) * 512 + lane * 16];
        v16bf b0 = *(const v16bf*)&bufB[(2 * wn + 0) * 512 + lane * 16];
        v16bf b1 = *(const v16bf*)&bufB[(2 * wn + 1) * 512 + lane * 16];
        acc[0][0] = __builtin_amdgcn_wmma_f32_16x16x32_bf16(false, a0, false, b0, (short)0, acc[0][0], false, false);
        acc[0][1] = __builtin_amdgcn_wmma_f32_16x16x32_bf16(false, a0, false, b1, (short)0, acc[0][1], false, false);
        acc[1][0] = __builtin_amdgcn_wmma_f32_16x16x32_bf16(false, a1, false, b0, (short)0, acc[1][0], false, false);
        acc[1][1] = __builtin_amdgcn_wmma_f32_16x16x32_bf16(false, a1, false, b1, (short)0, acc[1][1], false, false);
    };

    // ---- software-pipelined main loop (last iteration peeled) -------------
    const int nk = K / BK;
    loadA(); loadB();
    storeA(ldsA[0]); storeB(ldsB[0]);
    int cur = 0;
    for (int ks = 0; ks < nk - 1; ++ks) {
        __syncthreads();
        loadA(); loadB();
        // Pin the prefetch above the compute: loads must issue before the
        // WMMA block so their latency is covered; waits land in store phase.
        __builtin_amdgcn_sched_barrier(0);
        wmma_step(ldsA[cur], ldsB[cur]);
        storeA(ldsA[cur ^ 1]); storeB(ldsB[cur ^ 1]);
        cur ^= 1;
    }
    __syncthreads();
    wmma_step(ldsA[cur], ldsB[cur]);

    // ---- epilogue: C/D layout (ISA 7.12.2): VGPR r -> row = r + 8*half ----
#pragma unroll
    for (int i = 0; i < 2; ++i) {
        int rowbase = m0 + wm * 32 + i * 16;
#pragma unroll
        for (int j = 0; j < 2; ++j) {
            int col = n0 + wn * 32 + j * 16 + l16;
            float bv = HAS_BIAS ? bias[col] : 0.f;
#pragma unroll
            for (int r = 0; r < 8; ++r) {
                int row = rowbase + r + 8 * half;
                long long roff = (long long)row * ldc + col;
                float val = acc[i][j][r] + bv;
                if (RES >= 1) val += res1_scale * res1[roff];
                if (RES >= 2) val += res2[roff];
                Cb[roff] = val;
            }
        }
    }
}

// ---------------------------------------------------------------------------
// In-place row softmax with logit scale.  One block per row; N = 4096.
// float4 I/O: each thread owns 16 contiguous elements (4 x b128).
// ---------------------------------------------------------------------------
__global__ __launch_bounds__(256)
void softmax_rows(float* __restrict__ S, float scale)
{
    __shared__ float red[256];
    const int tid = threadIdx.x;
    float4* p = (float4*)(S + (long long)blockIdx.x * 4096) + tid * 4;

    float4 v[4];
    float mx = -3.4e38f;
#pragma unroll
    for (int i = 0; i < 4; ++i) {
        float4 t = p[i];
        t.x *= scale; t.y *= scale; t.z *= scale; t.w *= scale;
        v[i] = t;
        mx = fmaxf(mx, fmaxf(fmaxf(t.x, t.y), fmaxf(t.z, t.w)));
    }
    red[tid] = mx; __syncthreads();
    for (int s = 128; s > 0; s >>= 1) {
        if (tid < s) red[tid] = fmaxf(red[tid], red[tid + s]);
        __syncthreads();
    }
    mx = red[0]; __syncthreads();

    float sum = 0.f;
#pragma unroll
    for (int i = 0; i < 4; ++i) {
        v[i].x = __expf(v[i].x - mx); v[i].y = __expf(v[i].y - mx);
        v[i].z = __expf(v[i].z - mx); v[i].w = __expf(v[i].w - mx);
        sum += v[i].x + v[i].y + v[i].z + v[i].w;
    }
    red[tid] = sum; __syncthreads();
    for (int s = 128; s > 0; s >>= 1) {
        if (tid < s) red[tid] += red[tid + s];
        __syncthreads();
    }
    float inv = 1.f / red[0];
#pragma unroll
    for (int i = 0; i < 4; ++i) {
        v[i].x *= inv; v[i].y *= inv; v[i].z *= inv; v[i].w *= inv;
        p[i] = v[i];
    }
}

// ---------------------------------------------------------------------------
// GroupNorm over (H,W,C/G) per (batch,group). One block per (b,g).
// ---------------------------------------------------------------------------
__global__ __launch_bounds__(256)
void group_norm_k(const float* __restrict__ x, const float* __restrict__ gamma,
                  const float* __restrict__ beta, float* __restrict__ out,
                  int NHW, int C, int groups)
{
    __shared__ float rs[256], rs2[256];
    const int b = blockIdx.x / groups;
    const int g = blockIdx.x % groups;
    const int cg = C / groups;              // 20
    const int cnt = NHW * cg;               // 81920
    const float* xb = x + (long long)b * NHW * C;
    float* ob = out + (long long)b * NHW * C;

    float s = 0.f, s2 = 0.f;
    for (int idx = threadIdx.x; idx < cnt; idx += 256) {
        int n = idx / cg, c = g * cg + idx % cg;
        float v = xb[(long long)n * C + c];
        s += v; s2 += v * v;
    }
    rs[threadIdx.x] = s; rs2[threadIdx.x] = s2; __syncthreads();
    for (int t = 128; t > 0; t >>= 1) {
        if (threadIdx.x < t) {
            rs[threadIdx.x]  += rs[threadIdx.x + t];
            rs2[threadIdx.x] += rs2[threadIdx.x + t];
        }
        __syncthreads();
    }
    float mean = rs[0] / (float)cnt;
    float var  = rs2[0] / (float)cnt - mean * mean;
    float inv  = rsqrtf(var + 1e-3f);
    for (int idx = threadIdx.x; idx < cnt; idx += 256) {
        int n = idx / cg, c = g * cg + idx % cg;
        float v = xb[(long long)n * C + c];
        ob[(long long)n * C + c] = (v - mean) * inv * gamma[c] + beta[c];
    }
}

// ---------------------------------------------------------------------------
// GEGLU gate: ff[m,j] = xp[m,j] * quick_gelu(xp[m,H+j]);  xp is [M,2H].
// float4 vectorized along j (H = 1280 is a multiple of 4).
// ---------------------------------------------------------------------------
__global__ void geglu_mul(const float* __restrict__ xp, float* __restrict__ ff,
                          long long total4, int H4)   // H4 = H/4
{
    long long i = (long long)blockIdx.x * blockDim.x + threadIdx.x;
    if (i >= total4) return;
    long long m = i / H4, j = i % H4;
    const float4* row = (const float4*)xp + m * (2 * H4);
    float4 a = row[j];
    float4 g = row[H4 + j];
    float4 r;
    r.x = a.x * (g.x / (1.f + __expf(-1.702f * g.x)));
    r.y = a.y * (g.y / (1.f + __expf(-1.702f * g.y)));
    r.z = a.z * (g.z / (1.f + __expf(-1.702f * g.z)));
    r.w = a.w * (g.w / (1.f + __expf(-1.702f * g.w)));
    ((float4*)ff)[i] = r;
}

// ---------------------------------------------------------------------------
extern "C" void kernel_launch(void* const* d_in, const int* in_sizes, int n_in,
                              void* d_out, int out_size, void* d_ws, size_t ws_size,
                              hipStream_t stream)
{
    (void)in_sizes; (void)n_in; (void)out_size; (void)ws_size;

    const float* x      = (const float*)d_in[0];
    const float* ctx    = (const float*)d_in[1];
    const float* sa_q_w = (const float*)d_in[2];   const float* sa_q_b = (const float*)d_in[3];
    const float* sa_k_w = (const float*)d_in[4];   const float* sa_k_b = (const float*)d_in[5];
    const float* sa_v_w = (const float*)d_in[6];   const float* sa_v_b = (const float*)d_in[7];
    const float* sa_p_w = (const float*)d_in[8];   const float* sa_p_b = (const float*)d_in[9];
    const float* ca_q_w = (const float*)d_in[10];  const float* ca_q_b = (const float*)d_in[11];
    const float* ca_k_w = (const float*)d_in[12];  const float* ca_k_b = (const float*)d_in[13];
    const float* ca_v_w = (const float*)d_in[14];  const float* ca_v_b = (const float*)d_in[15];
    const float* ca_p_w = (const float*)d_in[16];  const float* ca_p_b = (const float*)d_in[17];
    const float* gn_g   = (const float*)d_in[18];  const float* gn_b   = (const float*)d_in[19];
    const float* gg_w   = (const float*)d_in[20];  const float* gg_b   = (const float*)d_in[21];
    const float* de_w   = (const float*)d_in[22];  const float* de_b   = (const float*)d_in[23];
    float* out = (float*)d_out;

    const int Bn = 2, N = 4096, C = 320;
    const int M  = Bn * N;                       // 8192
    const long long BNC = (long long)Bn * N * C; // 2.62M floats
    const long long NC  = (long long)N * C;
    const long long NN  = (long long)N * N;

    float* ws = (float*)d_ws;
    float* q  = ws;
    float* k  = ws + 1 * BNC;
    float* v  = ws + 2 * BNC;
    float* o  = ws + 3 * BNC;
    float* x1 = ws + 4 * BNC;
    float* xn = ws + 5 * BNC;
    float* x2 = ws + 6 * BNC;
    float* S  = ws + 7 * BNC;                    // B*N*N fp32 scores (134 MB)
    float* xp = S;                               // alias: scores dead by FFN
    float* ff = S + (long long)M * (8 * C);      // still inside the S arena

    const float scale = 1.0f / sqrtf((float)C);
    dim3 blk(BDIM);
    auto grid = [](int n, int m, int z) { return dim3(n / BN, m / BM, z); };

    // ---------------- self-attention ----------------
    gemm_bf16_wmma<false, true, 0><<<grid(C, M, 1), blk, 0, stream>>>(
        x, sa_q_w, sa_q_b, nullptr, 0.f, nullptr, q, C, C, C, C, 0, 0, 0);
    gemm_bf16_wmma<false, true, 0><<<grid(C, M, 1), blk, 0, stream>>>(
        x, sa_k_w, sa_k_b, nullptr, 0.f, nullptr, k, C, C, C, C, 0, 0, 0);
    gemm_bf16_wmma<false, true, 0><<<grid(C, M, 1), blk, 0, stream>>>(
        x, sa_v_w, sa_v_b, nullptr, 0.f, nullptr, v, C, C, C, C, 0, 0, 0);
    // S = Q @ K^T   (batched over B)
    gemm_bf16_wmma<true, false, 0><<<grid(N, N, Bn), blk, 0, stream>>>(
        q, k, nullptr, nullptr, 0.f, nullptr, S, C, C, C, N, NC, NC, NN);
    softmax_rows<<<dim3(Bn * N), dim3(256), 0, stream>>>(S, scale);
    // O = P @ V
    gemm_bf16_wmma<false, false, 0><<<grid(C, N, Bn), blk, 0, stream>>>(
        S, v, nullptr, nullptr, 0.f, nullptr, o, N, N, C, C, NN, NC, NC);
    // x1 = O @ Wp + bp + 2*x
    gemm_bf16_wmma<false, true, 1><<<grid(C, M, 1), blk, 0, stream>>>(
        o, sa_p_w, sa_p_b, x, 2.0f, nullptr, x1, C, C, C, C, 0, 0, 0);

    // ---------------- group norm ----------------
    group_norm_k<<<dim3(Bn * 16), dim3(256), 0, stream>>>(x1, gn_g, gn_b, xn, N, C, 16);

    // ---------------- cross-attention ----------------
    gemm_bf16_wmma<false, true, 0><<<grid(C, M, 1), blk, 0, stream>>>(
        xn, ca_q_w, ca_q_b, nullptr, 0.f, nullptr, q, C, C, C, C, 0, 0, 0);
    gemm_bf16_wmma<false, true, 0><<<grid(C, M, 1), blk, 0, stream>>>(
        ctx, ca_k_w, ca_k_b, nullptr, 0.f, nullptr, k, C, C, C, C, 0, 0, 0);
    gemm_bf16_wmma<false, true, 0><<<grid(C, M, 1), blk, 0, stream>>>(
        ctx, ca_v_w, ca_v_b, nullptr, 0.f, nullptr, v, C, C, C, C, 0, 0, 0);
    gemm_bf16_wmma<true, false, 0><<<grid(N, N, Bn), blk, 0, stream>>>(
        q, k, nullptr, nullptr, 0.f, nullptr, S, C, C, C, N, NC, NC, NN);
    softmax_rows<<<dim3(Bn * N), dim3(256), 0, stream>>>(S, scale);
    gemm_bf16_wmma<false, false, 0><<<grid(C, N, Bn), blk, 0, stream>>>(
        S, v, nullptr, nullptr, 0.f, nullptr, o, N, N, C, C, NN, NC, NC);
    // x2 = O @ Wp + bp + xn + x1
    gemm_bf16_wmma<false, true, 2><<<grid(C, M, 1), blk, 0, stream>>>(
        o, ca_p_w, ca_p_b, xn, 1.0f, x1, x2, C, C, C, C, 0, 0, 0);

    // ---------------- GEGLU FFN ----------------
    gemm_bf16_wmma<false, true, 0><<<grid(8 * C, M, 1), blk, 0, stream>>>(
        x2, gg_w, gg_b, nullptr, 0.f, nullptr, xp, C, C, 8 * C, 8 * C, 0, 0, 0);
    {
        long long total4 = (long long)M * C;   // M * 4C / 4
        int nb = (int)((total4 + 255) / 256);
        geglu_mul<<<dim3(nb), dim3(256), 0, stream>>>(xp, ff, total4, C);
    }
    // out = ff @ Wd + bd + x2
    gemm_bf16_wmma<false, true, 1><<<grid(C, M, 1), blk, 0, stream>>>(
        ff, de_w, de_b, x2, 1.0f, nullptr, out, 4 * C, 4 * C, C, C, 0, 0, 0);
}
